// Transformer_5634997093315
// MI455X (gfx1250) — matrix operand order repaired and checked by reading, not compiled, not run on hardware
//
#include <hip/hip_runtime.h>
#include <hip/hip_bf16.h>

// ---------------------------------------------------------------------------
// CDNA5 (gfx1250, wave32) transformer forward.
// All GEMM operands live in bf16 (weights pre-transposed to [N,K] once per
// call); one batched WMMA GEMM kernel (v_wmma_f32_16x16x32_bf16) with 128x128
// block tiles, double-buffered LDS, and GLOBAL_LOAD_ASYNC_TO_LDS_B128 staging
// tracked by ASYNCcnt (with a plain-copy fallback if builtins are absent).
// ---------------------------------------------------------------------------

typedef __attribute__((ext_vector_type(16))) __bf16 v16bf;
typedef __attribute__((ext_vector_type(8)))  __bf16 bf16x8;
typedef __attribute__((ext_vector_type(8)))  float  v8f;
typedef __attribute__((ext_vector_type(4)))  int    v4i;

// model dims
#define Bdim  64
#define Tdim  256
#define Edim  256
#define Hdim  8
#define DHdim 256
#define FHdim 1024
#define Ldim  6
#define Vdim  65

// GEMM tile config: 128x128 block, K-step 32, 256 threads = 8 wave32 waves.
// Wave grid 4(m) x 2(n): each wave owns 32 rows x 64 cols = 2x4 WMMA tiles.
#define GBM 128
#define GBN 128
#define GBK 32
#define LDST 48      // padded LDS row stride (elements) -> 96B rows, 16B aligned

#define FLAG_RELU  1
#define FLAG_CBF16 2

#if defined(__has_builtin)
#if __has_builtin(__builtin_amdgcn_global_load_async_to_lds_b128) && \
    __has_builtin(__builtin_amdgcn_s_wait_asynccnt)
#define USE_ASYNC 1
#endif
#endif
#ifndef USE_ASYNC
#define USE_ASYNC 0
#endif

__device__ __forceinline__ __bf16 f2bf(float f) { return (__bf16)f; }

// copy one 16-byte chunk global -> LDS
__device__ __forceinline__ void cp16(const __bf16* src, __bf16* dst)
{
#if USE_ASYNC
    __builtin_amdgcn_global_load_async_to_lds_b128(
        (__attribute__((address_space(1))) v4i*)(__bf16*)src,
        (__attribute__((address_space(3))) v4i*)dst, 0, 0);
#else
    *(bf16x8*)dst = *(const bf16x8*)src;
#endif
}

// stage a 128x32 bf16 tile (rows contiguous in k) into LDS [row][k], 2 chunks/thread
__device__ __forceinline__ void stage_tile(const __bf16* __restrict__ g, long ld,
                                           int row0, int rowLimit, int k0,
                                           __bf16* s, int tid, bool interior)
{
#pragma unroll
    for (int st = 0; st < 2; ++st) {
        const int c   = tid + st * 256;      // 0..511
        const int row = c >> 2;              // 0..127
        const int kc  = (c & 3) * 8;         // 0,8,16,24
        __bf16* dst = s + row * LDST + kc;
        const __bf16* src = g + (long)(row0 + row) * ld + (k0 + kc);
        if (interior) {
            cp16(src, dst);
        } else if (row0 + row < rowLimit) {
#pragma unroll
            for (int j = 0; j < 8; ++j) dst[j] = src[j];
        }
        // out-of-range rows keep stale LDS; those columns are never stored
    }
}

__device__ __forceinline__ void pipe_sync(bool interior)
{
#if USE_ASYNC
    if (interior) __builtin_amdgcn_s_wait_asynccnt(0);
#endif
    __syncthreads();
}

// C[z][M,N] = act(A[z][M,K] @ B[z][N,K]^T + bias[z][N]) ; A,B bf16; C f32 or bf16
// per-batch base offsets: off = (z/zDiv)*Out + (z%zDiv)*In  (elements)
__global__ __launch_bounds__(256)
void gemm_bf16_wmma(const __bf16* __restrict__ A, const __bf16* __restrict__ B,
                    const float* __restrict__ bias, void* __restrict__ Cout,
                    int M, int N, int K, int lda, int ldb, int ldc,
                    int zDiv, long aOut, long aIn, long bOut, long bIn,
                    long cOut, long cIn, long biasStride, int flags)
{
    __shared__ __align__(16) __bf16 As[2][GBM * LDST];   // [m][k]
    __shared__ __align__(16) __bf16 Bs[2][GBN * LDST];   // [n][k]

    const int  z  = (int)blockIdx.z;
    const long zo = z / zDiv, zi = z % zDiv;
    const __bf16* Ab = A + zo * aOut + zi * aIn;
    const __bf16* Bb = B + zo * bOut + zi * bIn;
    const long    co = zo * cOut + zi * cIn;
    const float* biasb = bias ? (bias + (long)z * biasStride) : nullptr;

    const int m0 = blockIdx.y * GBM;
    const int n0 = blockIdx.x * GBN;
    const bool interior = (m0 + GBM <= M) && (n0 + GBN <= N);

    const int tid    = threadIdx.x;
    const int lane   = tid & 31;
    const int wave   = tid >> 5;
    const int laneLo = lane & 15;
    const int laneHi = lane >> 4;
    const int mw = wave & 3;     // wave's 32-row group within 128
    const int nw = wave >> 2;    // wave's 64-col group within 128

    v8f acc[2][4];
#pragma unroll
    for (int mi = 0; mi < 2; ++mi)
#pragma unroll
        for (int ni = 0; ni < 4; ++ni) acc[mi][ni] = v8f{};

    // prologue: stage K-step 0 into buffer 0
    stage_tile(Ab, lda, m0, M, 0, As[0], tid, interior);
    stage_tile(Bb, ldb, n0, N, 0, Bs[0], tid, interior);

    int it = 0;
    for (int k0 = 0; k0 < K; k0 += GBK, ++it) {
        const int cur = it & 1;
        pipe_sync(interior);   // staging of `cur` complete, visible to all

        if (k0 + GBK < K) {    // stage next step into the other buffer
            stage_tile(Ab, lda, m0, M, k0 + GBK, As[cur ^ 1], tid, interior);
            stage_tile(Bb, ldb, n0, N, k0 + GBK, Bs[cur ^ 1], tid, interior);
        }

        // ---- A fragments (16x32): lanes 0-15 rows m, K runs {0..7,16..23};
        //      lanes 16-31 same rows, K runs {8..15,24..31} ----
        v16bf af[2];
#pragma unroll
        for (int mi = 0; mi < 2; ++mi) {
            const __bf16* arow =
                &As[cur][(32 * mw + 16 * mi + laneLo) * LDST + (laneHi ? 8 : 0)];
            bf16x8 alo = *(const bf16x8*)(arow);
            bf16x8 ahi = *(const bf16x8*)(arow + 16);
#pragma unroll
            for (int i = 0; i < 8; ++i) { af[mi][i] = alo[i]; af[mi][8 + i] = ahi[i]; }
        }
        // ---- B fragments (32x16): lane n = col, K runs same as A ----
        v16bf bfm[4];
#pragma unroll
        for (int ni = 0; ni < 4; ++ni) {
            const __bf16* brow =
                &Bs[cur][(64 * nw + 16 * ni + laneLo) * LDST + laneHi * 8];
            bf16x8 blo = *(const bf16x8*)(brow);
            bf16x8 bhi = *(const bf16x8*)(brow + 16);
#pragma unroll
            for (int i = 0; i < 8; ++i) { bfm[ni][i] = blo[i]; bfm[ni][8 + i] = bhi[i]; }
        }

        // ---- 8 WMMAs per wave per K-step ----
#pragma unroll
        for (int mi = 0; mi < 2; ++mi)
#pragma unroll
            for (int ni = 0; ni < 4; ++ni)
                acc[mi][ni] = __builtin_amdgcn_wmma_f32_16x16x32_bf16(
                    false, af[mi], false, bfm[ni], (short)0, acc[mi][ni],
                    false, false);
    }

    // ---- store C ----
    float*  C32 = (float*)Cout + co;
    __bf16* C16 = (__bf16*)Cout + co;
    const bool cbf = (flags & FLAG_CBF16) != 0;
#pragma unroll
    for (int mi = 0; mi < 2; ++mi) {
#pragma unroll
        for (int ni = 0; ni < 4; ++ni) {
            const int gn = n0 + 64 * nw + 16 * ni + laneLo;
            if (gn < N) {
                const float bv = biasb ? biasb[gn] : 0.0f;
#pragma unroll
                for (int r = 0; r < 8; ++r) {
                    const int gm = m0 + 32 * mw + 16 * mi + laneHi * 8 + r;
                    if (gm < M) {
                        float vv = acc[mi][ni][r] + bv;
                        if (flags & FLAG_RELU) vv = vv > 0.0f ? vv : 0.0f;
                        if (cbf) C16[(long)gm * ldc + gn] = f2bf(vv);
                        else     C32[(long)gm * ldc + gn] = vv;
                    }
                }
            }
        }
    }
}

// dst[z][n][k] = (bf16) src[z][k][n]   (weight convert + transpose, once per call)
__global__ __launch_bounds__(256)
void transpose_conv_bf16(const float* __restrict__ src, __bf16* __restrict__ dst,
                         int K, int N, long total)
{
    const long idx = (long)blockIdx.x * 256 + threadIdx.x;
    if (idx >= total) return;
    const int  k = (int)(idx % K);
    const long t = idx / K;
    const int  n = (int)(t % N);
    const long zz = t / N;
    dst[idx] = f2bf(src[zz * (long)K * N + (long)k * N + n]);
}

// vt[((b*H+h)*DH+d)*T + t] = v[(b*T+t)*HD + h*DH + d]
__global__ __launch_bounds__(256)
void transpose_v_bf16(const __bf16* __restrict__ v, __bf16* __restrict__ vt)
{
    const long idx = (long)blockIdx.x * 256 + threadIdx.x;
    const int  t = (int)(idx % Tdim);
    long r = idx / Tdim;
    const int  d = (int)(r % DHdim); r /= DHdim;
    const int  h = (int)(r % Hdim);
    const int  b = (int)(r / Hdim);
    vt[idx] = v[((long)b * Tdim + t) * (Hdim * DHdim) + h * DHdim + d];
}

// x[m,:] = emb[tok[m],:] * sqrt(E)  (f32 + bf16 copies)
__global__ __launch_bounds__(256)
void embed_scale(const int* __restrict__ tok, const float* __restrict__ emb,
                 float* __restrict__ x, __bf16* __restrict__ xbf)
{
    const long m = blockIdx.x;
    const int  c = threadIdx.x;
    const float v = emb[(long)tok[m] * Edim + c] * 16.0f; // sqrt(256)
    x[m * Edim + c]   = v;
    xbf[m * Edim + c] = f2bf(v);
}

// causal softmax over rows of S [B*H*T, T] (f32 in), bf16 probabilities out
__global__ __launch_bounds__(256)
void softmax_causal_bf16(const float* __restrict__ S, __bf16* __restrict__ P,
                         float scale)
{
    const long row = blockIdx.x;
    const int  t   = (int)(row % Tdim);
    const int  c   = threadIdx.x;

    float v = S[row * Tdim + c] * scale;
    if (c > t) v = -1.0e9f;

    __shared__ float red[256];
    red[c] = v; __syncthreads();
#pragma unroll
    for (int st = 128; st > 0; st >>= 1) {
        if (c < st) red[c] = fmaxf(red[c], red[c + st]);
        __syncthreads();
    }
    const float mx = red[0]; __syncthreads();

    const float e = __expf(v - mx);
    red[c] = e; __syncthreads();
#pragma unroll
    for (int st = 128; st > 0; st >>= 1) {
        if (c < st) red[c] += red[c + st];
        __syncthreads();
    }
    P[row * Tdim + c] = f2bf(e * (1.0f / red[0]));
}

// y = LayerNorm(a + r)*g + b  -> f32 y and bf16 ybf (eps=1e-5, population var)
__global__ __launch_bounds__(256)
void add_layernorm(const float* __restrict__ a, const float* __restrict__ r,
                   const float* __restrict__ g, const float* __restrict__ b,
                   float* __restrict__ y, __bf16* __restrict__ ybf)
{
    const long row = blockIdx.x;
    const int  c   = threadIdx.x;  // Edim threads
    const float v = a[row * Edim + c] + r[row * Edim + c];

    __shared__ float red[256];
    red[c] = v; __syncthreads();
#pragma unroll
    for (int st = 128; st > 0; st >>= 1) {
        if (c < st) red[c] += red[c + st];
        __syncthreads();
    }
    const float mu = red[0] * (1.0f / Edim); __syncthreads();

    const float d = v - mu;
    red[c] = d * d; __syncthreads();
#pragma unroll
    for (int st = 128; st > 0; st >>= 1) {
        if (c < st) red[c] += red[c + st];
        __syncthreads();
    }
    const float var = red[0] * (1.0f / Edim);
    const float o = d * rsqrtf(var + 1e-5f) * g[c] + b[c];
    y[row * Edim + c]   = o;
    ybf[row * Edim + c] = f2bf(o);
}

// ---------------------------------------------------------------------------

static inline void launch_gemm(hipStream_t s, const __bf16* A, const __bf16* B,
                               const float* bias, void* C,
                               int M, int N, int K, int lda, int ldb, int ldc,
                               int nz, int zDiv,
                               long aOut, long aIn, long bOut, long bIn,
                               long cOut, long cIn, long biasStride, int flags)
{
    dim3 grid((N + GBN - 1) / GBN, (M + GBM - 1) / GBM, nz);
    gemm_bf16_wmma<<<grid, dim3(256), 0, s>>>(A, B, bias, C, M, N, K,
                                              lda, ldb, ldc, zDiv,
                                              aOut, aIn, bOut, bIn, cOut, cIn,
                                              biasStride, flags);
}

extern "C" void kernel_launch(void* const* d_in, const int* in_sizes, int n_in,
                              void* d_out, int out_size, void* d_ws, size_t ws_size,
                              hipStream_t stream)
{
    const int*   tokens = (const int*)  d_in[0];
    const float* emb    = (const float*)d_in[1];
    const float* Wk     = (const float*)d_in[2];
    const float* Wq     = (const float*)d_in[3];
    const float* Wv     = (const float*)d_in[4];
    const float* bk     = (const float*)d_in[5];
    const float* bq     = (const float*)d_in[6];
    const float* bv     = (const float*)d_in[7];
    const float* Wp     = (const float*)d_in[8];
    const float* bp     = (const float*)d_in[9];
    const float* ln_g   = (const float*)d_in[10];
    const float* ln_b   = (const float*)d_in[11];
    const float* W1     = (const float*)d_in[12];
    const float* b1     = (const float*)d_in[13];
    const float* W2     = (const float*)d_in[14];
    const float* b2     = (const float*)d_in[15];
    const float* Wc1    = (const float*)d_in[16];
    const float* bc1    = (const float*)d_in[17];
    const float* Wc2    = (const float*)d_in[18];
    const float* bc2    = (const float*)d_in[19];
    float* out = (float*)d_out;

    const long MT = (long)Bdim * Tdim;      // 16384 token rows
    const long HD = (long)Hdim * DHdim;     // 2048
    const long NSC = (long)Bdim * Hdim * Tdim * Tdim;
    const long NX  = MT * Edim;
    const long NQ  = MT * HD;
    const long NFF = MT * FHdim;
    const long NWQ = (long)Ldim * Hdim * Edim * DHdim;

    float* ws     = (float*)d_ws;
    float* scores = ws;                 // f32 [B,H,T,T]
    float* x      = scores + NSC;       // f32 [MT,E]
    float* mid    = x + NX;
    float* proj   = mid + NX;
    __bf16* bb    = (__bf16*)(proj + NX);
    __bf16* xbf   = bb; bb += NX;
    __bf16* midbf = bb; bb += NX;
    __bf16* kb    = bb; bb += NQ;       // [B,T,H,DH]
    __bf16* qb    = bb; bb += NQ;
    __bf16* vb    = bb; bb += NQ;
    __bf16* vt    = bb; bb += NQ;       // [B,H,DH,T]
    __bf16* attnb = bb; bb += NQ;       // [B,H,T,T] bf16 probs
    __bf16* aoutb = bb; bb += NQ;       // [MT,HD]
    __bf16* ffnb  = bb; bb += NFF;      // [MT,FH]
    __bf16* wkt   = bb; bb += NWQ;      // [L,H,DH,E]
    __bf16* wqt   = bb; bb += NWQ;
    __bf16* wvt   = bb; bb += NWQ;
    __bf16* wpt   = bb; bb += (long)Ldim * HD * Edim;    // [L,E,HD]
    __bf16* w1t   = bb; bb += (long)Ldim * Edim * FHdim; // [L,FH,E]
    __bf16* w2t   = bb; bb += (long)Ldim * FHdim * Edim; // [L,E,FH]
    __bf16* wc1t  = bb; bb += (long)Edim * FHdim;        // [FH,E]
    __bf16* wc2t  = bb;                                  // [V,FH]

    // weight convert + transpose (once per call)
    auto conv = [&](const float* s_, __bf16* d_, int Kk, int Nn, long zc) {
        const long tot = zc * (long)Kk * Nn;
        transpose_conv_bf16<<<dim3((unsigned)((tot + 255) / 256)), dim3(256), 0,
                              stream>>>(s_, d_, Kk, Nn, tot);
    };
    conv(Wk,  wkt,  Edim, DHdim, (long)Ldim * Hdim);
    conv(Wq,  wqt,  Edim, DHdim, (long)Ldim * Hdim);
    conv(Wv,  wvt,  Edim, DHdim, (long)Ldim * Hdim);
    conv(Wp,  wpt,  (int)HD, Edim, Ldim);
    conv(W1,  w1t,  Edim, FHdim, Ldim);
    conv(W2,  w2t,  FHdim, Edim, Ldim);
    conv(Wc1, wc1t, Edim, FHdim, 1);
    conv(Wc2, wc2t, FHdim, Vdim, 1);

    embed_scale<<<dim3((unsigned)MT), dim3(256), 0, stream>>>(tokens, emb, x, xbf);

    for (int i = 0; i < Ldim; ++i) {
        // K,Q,V projections (z = h): C[b*T+t, h*DH+n] bf16
        launch_gemm(stream, xbf, wkt + (long)i * Hdim * DHdim * Edim,
                    bk + (long)i * Hdim * DHdim, kb,
                    (int)MT, DHdim, Edim, Edim, Edim, (int)HD, Hdim, 1,
                    0L, 0L, (long)DHdim * Edim, 0L, (long)DHdim, 0L,
                    DHdim, FLAG_CBF16);
        launch_gemm(stream, xbf, wqt + (long)i * Hdim * DHdim * Edim,
                    bq + (long)i * Hdim * DHdim, qb,
                    (int)MT, DHdim, Edim, Edim, Edim, (int)HD, Hdim, 1,
                    0L, 0L, (long)DHdim * Edim, 0L, (long)DHdim, 0L,
                    DHdim, FLAG_CBF16);
        launch_gemm(stream, xbf, wvt + (long)i * Hdim * DHdim * Edim,
                    bv + (long)i * Hdim * DHdim, vb,
                    (int)MT, DHdim, Edim, Edim, Edim, (int)HD, Hdim, 1,
                    0L, 0L, (long)DHdim * Edim, 0L, (long)DHdim, 0L,
                    DHdim, FLAG_CBF16);

        // scores[b,h,t,s] = sum_d K[b,t,h,d]*Q[b,s,h,d]   (z = b*H + h)
        launch_gemm(stream, kb, qb, nullptr, scores, Tdim, Tdim, DHdim,
                    (int)HD, (int)HD, Tdim, Bdim * Hdim, Hdim,
                    (long)Tdim * HD, (long)DHdim,
                    (long)Tdim * HD, (long)DHdim,
                    (long)Hdim * Tdim * Tdim, (long)Tdim * Tdim,
                    0L, 0);

        // scale + causal mask + softmax -> bf16 probs
        softmax_causal_bf16<<<dim3((unsigned)(Bdim * Hdim * Tdim)), dim3(256), 0,
                              stream>>>(scores, attnb, 0.0625f);

        // V -> [B,H,DH,T] for [N,K]-style B operand
        transpose_v_bf16<<<dim3((unsigned)((NQ + 255) / 256)), dim3(256), 0,
                           stream>>>(vb, vt);

        // out[b,t,h,d] = sum_s attn[b,h,t,s] * V[b,s,h,d]
        launch_gemm(stream, attnb, vt, nullptr, aoutb, Tdim, DHdim, Tdim,
                    Tdim, Tdim, (int)HD, Bdim * Hdim, Hdim,
                    (long)Hdim * Tdim * Tdim, (long)Tdim * Tdim,
                    (long)Hdim * DHdim * Tdim, (long)DHdim * Tdim,
                    (long)Tdim * HD, (long)DHdim,
                    0L, FLAG_CBF16);

        // proj = aout @ Wp + bp   (f32 out)
        launch_gemm(stream, aoutb, wpt + (long)i * Edim * HD,
                    bp + (long)i * Edim, proj,
                    (int)MT, Edim, (int)HD, (int)HD, (int)HD, Edim,
                    1, 1, 0L, 0L, 0L, 0L, 0L, 0L, 0L, 0);

        // mid = LN(x + proj)
        add_layernorm<<<dim3((unsigned)MT), dim3(256), 0, stream>>>(
            x, proj, ln_g + (long)i * Edim, ln_b + (long)i * Edim, mid, midbf);

        // ffn = relu(mid @ W1 + b1)  (bf16 out)
        launch_gemm(stream, midbf, w1t + (long)i * FHdim * Edim,
                    b1 + (long)i * FHdim, ffnb,
                    (int)MT, FHdim, Edim, Edim, Edim, FHdim,
                    1, 1, 0L, 0L, 0L, 0L, 0L, 0L, 0L, FLAG_RELU | FLAG_CBF16);

        // proj = ffn @ W2 + b2  (f32 out)
        launch_gemm(stream, ffnb, w2t + (long)i * Edim * FHdim,
                    b2 + (long)i * Edim, proj,
                    (int)MT, Edim, FHdim, FHdim, FHdim, Edim,
                    1, 1, 0L, 0L, 0L, 0L, 0L, 0L, 0L, 0);

        // x = LN(mid + proj)
        add_layernorm<<<dim3((unsigned)MT), dim3(256), 0, stream>>>(
            mid, proj, ln_g + (long)i * Edim, ln_b + (long)i * Edim, x, xbf);
    }

    // classifier: logits = relu(x @ Wc1 + bc1) @ Wc2 + bc2
    launch_gemm(stream, xbf, wc1t, bc1, ffnb, (int)MT, FHdim, Edim,
                Edim, Edim, FHdim, 1, 1, 0L, 0L, 0L, 0L, 0L, 0L, 0L,
                FLAG_RELU | FLAG_CBF16);
    launch_gemm(stream, ffnb, wc2t, bc2, out, (int)MT, Vdim, FHdim,
                FHdim, FHdim, Vdim, 1, 1, 0L, 0L, 0L, 0L, 0L, 0L, 0L, 0);
}
